// GNNLayer_77120432767347
// MI455X (gfx1250) — matrix operand order repaired and checked by reading, not compiled, not run on hardware
//
#include <hip/hip_runtime.h>

// ---------------- problem constants (match reference) ----------------
#define ND      10000      // N_DRUGS
#define NT      20000      // N_TAILS
#define NR      100        // N_RELS
#define E_EDGES 1000000
#define DIM     64
#define SSZ     16         // SAMPLE_SIZE
#define NTILE_E (E_EDGES / 16)   // 62500
#define NTILE_D (ND / 16)        // 625
#define EDGE_BLOCKS 1250         // persistent-wave grid for the edge kernel

// ---------------- WMMA vector types ----------------
typedef __attribute__((ext_vector_type(16))) __bf16 v16bf;
typedef __attribute__((ext_vector_type(8)))  float  v8f;

static __device__ __forceinline__ unsigned hashu(unsigned x) {
    x ^= x >> 16; x *= 0x7feb352dU;
    x ^= x >> 15; x *= 0x846ca68bU;
    x ^= x >> 16; return x;
}

static __device__ __forceinline__ unsigned long long shflxor_u64(unsigned long long x, int m) {
    unsigned lo = (unsigned)x, hi = (unsigned)(x >> 32);
    lo = __shfl_xor(lo, m, 32);
    hi = __shfl_xor(hi, m, 32);
    return ((unsigned long long)hi << 32) | (unsigned long long)lo;
}

// ---------------- tiny setup: w2sum[k] = sum_n W2[k][n]; w2sum[64] = sum(b2) ----------------
__global__ void k_w2sum(const float* __restrict__ W2, const float* __restrict__ b2,
                        float* __restrict__ w2s) {
    const int k = threadIdx.x;
    if (k < DIM) {
        float s = 0.f;
        for (int n = 0; n < DIM; ++n) s += W2[k * DIM + n];
        w2s[k] = s;
    }
    if (k == 0) {
        float s = 0.f;
        for (int n = 0; n < DIM; ++n) s += b2[n];
        w2s[DIM] = s;
    }
}

__global__ void k_zero32(unsigned* __restrict__ p, int n) {
    const int i = blockIdx.x * blockDim.x + threadIdx.x;
    if (i < n) p[i] = 0u;
}

// ---------------- edge MLP: score[e] = sum_n( sigmoid(h@W1+b1) @ W2 + b2 ) ----------------
// = sigmoid(h@W1+b1) . rowsum(W2) + sum(b2).
// Persistent waves: B-tiles of W1 (8 x v16bf = 64 VGPRs) + b1/w2sum lane values are
// loaded ONCE per wave, then the wave grid-strides over 16-edge tiles.
__global__ void __launch_bounds__(256) k_edge_mlp(
    const float* __restrict__ drug_emb, const float* __restrict__ rel_emb,
    const float* __restrict__ W1, const float* __restrict__ b1,
    const float* __restrict__ w2s, const int* __restrict__ DKG,
    float* __restrict__ score)
{
    __shared__ __bf16 w1bf[DIM * DIM];   // 8 KB bf16 copy of W1
    for (int i = threadIdx.x; i < DIM * DIM; i += blockDim.x) w1bf[i] = (__bf16)W1[i];
    __syncthreads();

    const int wave = threadIdx.x >> 5, lane = threadIdx.x & 31;
    const int half = lane >> 4, m = lane & 15;
    const int nwaves = EDGE_BLOCKS * 8;

    // ---- loop-invariant operands ----
    // B-matrix (32x16 bf16): lanes 0-15 hold K=0..15, lanes 16-31 hold K=16..31.
    v16bf bt[4][2];
    float b1v[4], wsv[4];
#pragma unroll
    for (int nb = 0; nb < 4; ++nb) {
        const int ncol = nb * 16 + m;
#pragma unroll
        for (int kb = 0; kb < 2; ++kb) {
#pragma unroll
            for (int j = 0; j < 16; ++j)
                bt[nb][kb][j] = w1bf[(kb * 32 + half * 16 + j) * DIM + ncol];
        }
        b1v[nb] = b1[ncol];
        wsv[nb] = w2s[ncol];
    }
    const float b2sum = w2s[DIM];

    for (int tile = blockIdx.x * 8 + wave; tile < NTILE_E; tile += nwaves) {
        const int e0 = tile * 16;
        const int e  = e0 + m;

        // prefetch next tile's DKG rows (speculative, dropped if OOB)
        __builtin_prefetch(&DKG[(e + nwaves * 16) * 3], 0, 1);

        const int hd = DKG[e * 3 + 0];
        const int rl = DKG[e * 3 + 2];
        const float* de = drug_emb + hd * DIM;
        const float* re = rel_emb  + rl * DIM;

        // A-matrix (16x32 bf16): lanes 0-15 -> M=lane; element j -> K=16*half+j (same order as B)
        v16bf a0, a1;
#pragma unroll
        for (int j = 0; j < 16; ++j) {
            const int k = half * 16 + j;
            a0[j] = (__bf16)(de[k]      * re[k]);
            a1[j] = (__bf16)(de[k + 32] * re[k + 32]);
        }

        float tacc[8];
#pragma unroll
        for (int v = 0; v < 8; ++v) tacc[v] = 0.f;

#pragma unroll
        for (int nb = 0; nb < 4; ++nb) {
            v8f c;
#pragma unroll
            for (int v = 0; v < 8; ++v) c[v] = b1v[nb];
            c = __builtin_amdgcn_wmma_f32_16x16x32_bf16(false, a0, false, bt[nb][0], (short)0, c, false, false);
            c = __builtin_amdgcn_wmma_f32_16x16x32_bf16(false, a1, false, bt[nb][1], (short)0, c, false, false);
            // fused 2nd layer + rowsum: sigmoid(c) . w2sum
#pragma unroll
            for (int v = 0; v < 8; ++v) {
                const float sg = 1.f / (1.f + __expf(-c[v]));
                tacc[v] += sg * wsv[nb];
            }
        }
        // reduce over the 16 lanes of each half (masks < 16 stay within halves)
#pragma unroll
        for (int v = 0; v < 8; ++v) {
#pragma unroll
            for (int msk = 1; msk < 16; msk <<= 1)
                tacc[v] += __shfl_xor(tacc[v], msk, 32);
        }
        if (m == 0) {
#pragma unroll
            for (int v = 0; v < 8; ++v)
                score[e0 + half * 8 + v] = tacc[v] + b2sum;
        }
    }
}

// ---------------- CSR build ----------------
__global__ void k_deg(const int* __restrict__ DKG, int* __restrict__ deg) {
    const int i = blockIdx.x * blockDim.x + threadIdx.x;
    if (i < E_EDGES) atomicAdd(&deg[DKG[i * 3]], 1);
}

__global__ void __launch_bounds__(256) k_scan(const int* __restrict__ deg, int* __restrict__ offs) {
    __shared__ int part[256];
    const int t = threadIdx.x;
    const int chunk = (ND + 255) / 256;
    const int lo = t * chunk;
    int hi = lo + chunk; if (hi > ND) hi = ND;
    int s = 0;
    for (int i = lo; i < hi; ++i) s += deg[i];
    part[t] = s;
    __syncthreads();
    if (t == 0) {
        int acc = 0;
        for (int i = 0; i < 256; ++i) { const int p = part[i]; part[i] = acc; acc += p; }
    }
    __syncthreads();
    int acc = part[t];
    for (int i = lo; i < hi; ++i) { offs[i] = acc; acc += deg[i]; }
}

__global__ void k_scatter(const int* __restrict__ DKG, const int* __restrict__ offs,
                          int* __restrict__ curs, int* __restrict__ csre,
                          unsigned* __restrict__ csrk) {
    const int i = blockIdx.x * blockDim.x + threadIdx.x;
    if (i < E_EDGES) {
        const int h = DKG[i * 3];
        const int pos = offs[h] + atomicAdd(&curs[h], 1);
        csre[pos] = i;
        csrk[pos] = hashu((unsigned)i * 0x9E3779B1u + 0x85EBCA77u);
    }
}

// ---------------- per-head sample + aggregate: neigh[d] = sum score[e]*tail_emb[tail[e]] ----------------
// One wave per head. Deterministic: 16 rounds of argmin over (key<<32|edge) ranks.
__global__ void __launch_bounds__(256) k_select_agg(
    const int* __restrict__ offs, const int* __restrict__ deg,
    const int* __restrict__ csre, const unsigned* __restrict__ csrk,
    const float* __restrict__ score, const float* __restrict__ tail_emb,
    const int* __restrict__ DKG, float* __restrict__ neigh)
{
    const int wave = threadIdx.x >> 5, lane = threadIdx.x & 31;
    const int d = blockIdx.x * 8 + wave;
    if (d >= ND) return;
    const int start = offs[d];
    const int dg    = deg[d];
    const int k2    = lane * 2;          // each lane owns 2 of the 64 dims
    if (dg == 0) {
        neigh[d * DIM + k2] = 0.f; neigh[d * DIM + k2 + 1] = 0.f;
        return;
    }
    int sel[SSZ];
    const int rounds = dg < SSZ ? dg : SSZ;
    unsigned long long last = 0ull;
    for (int i = 0; i < rounds; ++i) {
        unsigned long long best = ~0ull;
        for (int t = lane; t < dg; t += 32) {
            const unsigned eid = (unsigned)csre[start + t];
            const unsigned long long rk =
                ((unsigned long long)csrk[start + t] << 32) | (unsigned long long)(eid + 1u);
            if (rk > last && rk < best) best = rk;
        }
#pragma unroll
        for (int msk = 1; msk < 32; msk <<= 1) {
            const unsigned long long o = shflxor_u64(best, msk);
            if (o < best) best = o;
        }
        sel[i] = (int)((unsigned)best) - 1;   // all lanes hold the winner
        last = best;
    }
    // deg < S: extra draws WITH replacement from the rank-sorted list (deterministic)
    for (int i = rounds; i < SSZ; ++i) {
        const unsigned u = hashu((unsigned)d * 0x01000193u ^ (unsigned)(i * 0x9E3779B9)) % (unsigned)dg;
        sel[i] = sel[u];
    }
    float a0 = 0.f, a1 = 0.f;
    for (int i = 0; i < SSZ; ++i) {
        const int e = sel[i];
        const int t = DKG[e * 3 + 1];
        const float s = score[e];
        a0 += s * tail_emb[t * DIM + k2];
        a1 += s * tail_emb[t * DIM + k2 + 1];
    }
    neigh[d * DIM + k2]     = a0;
    neigh[d * DIM + k2 + 1] = a1;
}

// ---------------- combine: y = [drug_emb | neigh] @ Wc + bc (10000x128 @ 128x64) ----------------
__global__ void __launch_bounds__(256) k_combine(
    const float* __restrict__ drug_emb, const float* __restrict__ neigh,
    const float* __restrict__ Wc, const float* __restrict__ bc,
    float* __restrict__ y)
{
    __shared__ __bf16 wcbf[2 * DIM * DIM];   // 16 KB bf16 copy of Wc
    for (int i = threadIdx.x; i < 2 * DIM * DIM; i += blockDim.x) wcbf[i] = (__bf16)Wc[i];
    __syncthreads();

    const int wave = threadIdx.x >> 5, lane = threadIdx.x & 31;
    const int tile = blockIdx.x * 8 + wave;
    if (tile >= NTILE_D) return;
    const int r0   = tile * 16;
    const int half = lane >> 4, m = lane & 15;
    const int row  = r0 + m;
    const float* dr = drug_emb + row * DIM;
    const float* nr = neigh    + row * DIM;

    v16bf a[4];
#pragma unroll
    for (int kb = 0; kb < 4; ++kb) {
#pragma unroll
        for (int j = 0; j < 16; ++j) {
            const int kl = half * 16 + j;
            const int kg = kb * 32 + kl;
            const float val = (kg < DIM) ? dr[kg] : nr[kg - DIM];
            a[kb][j] = (__bf16)val;
        }
    }
#pragma unroll
    for (int nb = 0; nb < 4; ++nb) {
        const int ncol = nb * 16 + m;
        v8f c;
        const float bb = bc[ncol];
#pragma unroll
        for (int v = 0; v < 8; ++v) c[v] = bb;
#pragma unroll
        for (int kb = 0; kb < 4; ++kb) {
            v16bf bt;
#pragma unroll
            for (int j = 0; j < 16; ++j) {
                const int kk = kb * 32 + half * 16 + j;
                bt[j] = wcbf[kk * DIM + ncol];
            }
            c = __builtin_amdgcn_wmma_f32_16x16x32_bf16(false, a[kb], false, bt, (short)0, c, false, false);
        }
#pragma unroll
        for (int v = 0; v < 8; ++v)
            y[(r0 + half * 8 + v) * DIM + ncol] = c[v];
    }
}

// ---------------- batch-norm stats: deterministic per-column block reduction ----------------
__global__ void __launch_bounds__(256) k_colreduce(const float* __restrict__ y,
                                                   float* __restrict__ colsum,
                                                   float* __restrict__ colsq) {
    __shared__ float ss[256], sq[256];
    const int col = blockIdx.x;    // 64 blocks
    float s = 0.f, q = 0.f;
    for (int r = threadIdx.x; r < ND; r += 256) {
        const float v = y[r * DIM + col];
        s += v; q += v * v;
    }
    ss[threadIdx.x] = s; sq[threadIdx.x] = q;
    __syncthreads();
    for (int off = 128; off > 0; off >>= 1) {
        if (threadIdx.x < off) {
            ss[threadIdx.x] += ss[threadIdx.x + off];
            sq[threadIdx.x] += sq[threadIdx.x + off];
        }
        __syncthreads();
    }
    if (threadIdx.x == 0) { colsum[col] = ss[0]; colsq[col] = sq[0]; }
}

__global__ void k_bn(float* __restrict__ y, const float* __restrict__ colsum,
                     const float* __restrict__ colsq,
                     const float* __restrict__ gamma, const float* __restrict__ beta) {
    const int i = blockIdx.x * blockDim.x + threadIdx.x;
    if (i >= ND * DIM) return;
    const int c = i & (DIM - 1);
    const float mean = colsum[c] * (1.0f / ND);
    const float var  = colsq[c]  * (1.0f / ND) - mean * mean;
    y[i] = gamma[c] * (y[i] - mean) * rsqrtf(var + 1e-5f) + beta[c];
}

// ---------------- host-side launch ----------------
extern "C" void kernel_launch(void* const* d_in, const int* in_sizes, int n_in,
                              void* d_out, int out_size, void* d_ws, size_t ws_size,
                              hipStream_t stream)
{
    (void)in_sizes; (void)n_in; (void)out_size; (void)ws_size;
    const float* HF    = (const float*)d_in[0];
    const float* X     = (const float*)d_in[1];
    const float* drug  = (const float*)d_in[2];
    const float* rel   = (const float*)d_in[3];
    const float* tail  = (const float*)d_in[4];
    const float* W1    = (const float*)d_in[5];
    const float* b1    = (const float*)d_in[6];
    const float* W2    = (const float*)d_in[7];
    const float* b2    = (const float*)d_in[8];
    const float* Wc    = (const float*)d_in[9];
    const float* bc    = (const float*)d_in[10];
    const float* gamma = (const float*)d_in[11];
    const float* beta  = (const float*)d_in[12];
    const int*   DKG   = (const int*)d_in[13];

    float* out   = (float*)d_out;
    float* outHF = out;                 // 10000*64
    float* outY  = out + ND * DIM;      // 10000*64
    float* outX  = out + 2 * ND * DIM;  // 10000

    // workspace carve (~14.7 MB)
    float*    score  = (float*)d_ws;                 // E
    float*    w2s    = score + E_EDGES;              // 65 (padded 128)
    int*      deg    = (int*)(w2s + 128);            // ND
    int*      offs   = deg + ND;                     // ND
    int*      curs   = offs + ND;                    // ND
    int*      csre   = curs + ND;                    // E
    unsigned* csrk   = (unsigned*)(csre + E_EDGES);  // E
    float*    neigh  = (float*)(csrk + E_EDGES);     // ND*DIM
    float*    colsum = neigh + ND * DIM;             // 64
    float*    colsq  = colsum + DIM;                 // 64

    // passthrough outputs (tuple elements 0 and 2)
    hipMemcpyAsync(outHF, HF, sizeof(float) * ND * DIM, hipMemcpyDeviceToDevice, stream);
    hipMemcpyAsync(outX,  X,  sizeof(float) * ND,       hipMemcpyDeviceToDevice, stream);

    // zero deg/offs/curs (contiguous)
    k_zero32<<<(3 * ND + 255) / 256, 256, 0, stream>>>((unsigned*)deg, 3 * ND);
    k_w2sum<<<1, 64, 0, stream>>>(W2, b2, w2s);
    k_edge_mlp<<<EDGE_BLOCKS, 256, 0, stream>>>(drug, rel, W1, b1, w2s, DKG, score);
    k_deg<<<(E_EDGES + 255) / 256, 256, 0, stream>>>(DKG, deg);
    k_scan<<<1, 256, 0, stream>>>(deg, offs);
    k_scatter<<<(E_EDGES + 255) / 256, 256, 0, stream>>>(DKG, offs, curs, csre, csrk);
    k_select_agg<<<(ND + 7) / 8, 256, 0, stream>>>(offs, deg, csre, csrk, score, tail, DKG, neigh);
    k_combine<<<(NTILE_D + 7) / 8, 256, 0, stream>>>(drug, neigh, Wc, bc, outY);
    k_colreduce<<<DIM, 256, 0, stream>>>(outY, colsum, colsq);
    k_bn<<<(ND * DIM + 255) / 256, 256, 0, stream>>>(outY, colsum, colsq, gamma, beta);
}
